// MultiHeadAttention_12120397709600
// MI455X (gfx1250) — matrix operand order repaired and checked by reading, not compiled, not run on hardware
//
#include <hip/hip_runtime.h>

// ---------------------------------------------------------------------------
// MI455X (gfx1250, wave32) fused MHA:
//   1) cvt weights + activations f32->bf16 (vectorized, bandwidth-bound)
//   2) WMMA GEMM  Q = (query @ Wq.T)*(log2e/sqrt(dk)), K = key @ Wk.T,
//                 Vt = (value @ Wv.T)^T          (pure load+wmma inner loop)
//   3) flash-attention: exp2-domain online softmax (raw v_exp), WMMA row-sums,
//      global_prefetch of next K/V block
//   4) WMMA GEMM  out = ctx @ Wout.T + b_out    (fp32 output)
// ---------------------------------------------------------------------------

typedef unsigned short u16;
typedef unsigned int   u32;
typedef __attribute__((ext_vector_type(16))) __bf16 v16bf;
typedef __attribute__((ext_vector_type(8)))  float  v8f;
typedef __attribute__((ext_vector_type(4)))  u32    v4u;
typedef __attribute__((ext_vector_type(4)))  float  v4f;

union BF16Frag { v4u q[2]; v16bf v; };          // 16 bf16 = 32B
union BF16Build { u16 u[16]; v16bf v; };
union Pack8 { u16 u[8]; v4u q; };

__device__ __forceinline__ u16 f2bf(float f) {
  u32 u = __float_as_uint(f);
  u = (u + 0x7fffu + ((u >> 16) & 1u)) >> 16;   // round-to-nearest-even
  return (u16)u;
}

__device__ __forceinline__ u32 f2bf_pk(float lo, float hi) {
  return (u32)f2bf(lo) | ((u32)f2bf(hi) << 16);
}

__device__ __forceinline__ v16bf ones_bf16() {
  BF16Build b;
#pragma unroll
  for (int i = 0; i < 16; ++i) b.u[i] = 0x3F80;  // 1.0 bf16
  return b.v;
}

// raw v_exp_f32 (exp2). Args are always <= 0 here; flush-to-zero is desired.
__device__ __forceinline__ float fast_exp2(float x) {
#if __has_builtin(__builtin_amdgcn_exp2f)
  return __builtin_amdgcn_exp2f(x);
#else
  return exp2f(x);
#endif
}

__device__ __forceinline__ float fast_rcp(float x) {
#if __has_builtin(__builtin_amdgcn_rcpf)
  return __builtin_amdgcn_rcpf(x);
#else
  return 1.0f / x;
#endif
}

static constexpr int S_LEN = 2048;
static constexpr int DMODEL = 1024;
static constexpr int NHEAD = 16;
static constexpr int DK = 64;
static constexpr int BATCH = 2;
static constexpr float LOG2E = 1.4426950408889634f;
static constexpr float NEG_BIG = -1.0e30f;

// ---------------------------------------------------------------------------
// 8 floats -> 8 packed bf16 per thread (16B load x2 -> 16B store)
__global__ void cvt_f32_bf16(const float* __restrict__ s, u16* __restrict__ d, int n8) {
  int i = blockIdx.x * blockDim.x + threadIdx.x;
  int stride = gridDim.x * blockDim.x;
  for (; i < n8; i += stride) {
    v4f a = *(const v4f*)(s + (size_t)i * 8);
    v4f b = *(const v4f*)(s + (size_t)i * 8 + 4);
    v4u o;
    o.x = f2bf_pk(a.x, a.y);
    o.y = f2bf_pk(a.z, a.w);
    o.z = f2bf_pk(b.x, b.y);
    o.w = f2bf_pk(b.z, b.w);
    *(v4u*)(d + (size_t)i * 8) = o;
  }
}

// ---------------------------------------------------------------------------
// O[m,n] = sum_k A[m,k] * W[n,k]   (A bf16 MxK row-major, W bf16 NxK row-major)
// One wave -> 16x64 output strip. K = N = 1024, M = 4096.
// MODE 0: Q store (scale log2e/8, bf16 [B,H,S,dk])  MODE 1: K store (bf16 same)
// MODE 2: V store transposed (bf16 [B,H,dk,S])      MODE 3: fp32 + bias to d_out
template <int MODE>
__global__ __launch_bounds__(256) void gemm_xwt(
    const u16* __restrict__ A, const u16* __restrict__ W,
    u16* __restrict__ obf, float* __restrict__ of, const float* __restrict__ bias) {
  const int Kd = DMODEL;
  const int lane = threadIdx.x & 31;
  const int wid = blockIdx.x * (blockDim.x >> 5) + (threadIdx.x >> 5);
  const int mtile = wid >> 4;          // 0..255  (M/16)
  const int nstrip = wid & 15;         // 0..15   (N/64)  == head index
  const int m0 = mtile * 16;
  const int n0s = nstrip * 64;
  const bool hi = lane >= 16;
  const int lm = lane & 15;

  v8f acc[4] = {};
  const u16* arow = A + (size_t)(m0 + lm) * Kd;

  for (int kb = 0; kb < Kd; kb += 32) {
    // A fragment (16x32 bf16): lane<16 -> K {0..15}, lane>=16 -> {16..31}
    BF16Frag af;
    af.q[0] = *(const v4u*)(arow + kb + (hi ? 16 : 0));
    af.q[1] = *(const v4u*)(arow + kb + (hi ? 24 : 8));

    // B fragments: lane = column n, contiguous K halves (0..15 / 16..31)
    const int ks = kb + (hi ? 16 : 0);
#pragma unroll
    for (int t = 0; t < 4; ++t) {
      const u16* wrow = W + (size_t)(n0s + t * 16 + lm) * Kd + ks;
      BF16Frag bf;
      bf.q[0] = *(const v4u*)(wrow);
      bf.q[1] = *(const v4u*)(wrow + 8);
      acc[t] = __builtin_amdgcn_wmma_f32_16x16x32_bf16(
          false, af.v, false, bf.v, (short)0, acc[t], false, false);
    }
  }

  const int b = m0 >> 11;              // m0 / 2048
  const int sb = m0 & (S_LEN - 1);
  const int rbase = hi ? 8 : 0;

  if (MODE == 0 || MODE == 1) {
    // Q gets log2e/sqrt(dk) so attention scores land in the exp2 domain
    const float scale = (MODE == 0) ? (0.125f * LOG2E) : 1.0f;
    const int h = nstrip;
#pragma unroll
    for (int t = 0; t < 4; ++t) {
      const int off = t * 16 + lm;
      u16* base = obf + ((size_t)(b * NHEAD + h) * S_LEN) * DK + off;
#pragma unroll
      for (int r = 0; r < 8; ++r) {
        const int s = sb + r + rbase;
        base[(size_t)s * DK] = f2bf(acc[t][r] * scale);
      }
    }
  } else if (MODE == 2) {
    const int h = nstrip;
#pragma unroll
    for (int t = 0; t < 4; ++t) {
      const int off = t * 16 + lm;
      Pack8 pk;
#pragma unroll
      for (int r = 0; r < 8; ++r) pk.u[r] = f2bf(acc[t][r]);
      u16* dst = obf + ((size_t)(b * NHEAD + h) * DK + off) * S_LEN + sb + rbase;
      *(v4u*)dst = pk.q;               // 8 consecutive s positions, 16B store
    }
  } else {
#pragma unroll
    for (int t = 0; t < 4; ++t) {
      const int n = n0s + t * 16 + lm;
      const float bv = bias[n];
      float* dst = of + (size_t)m0 * DMODEL + n;
#pragma unroll
      for (int r = 0; r < 8; ++r)
        dst[(size_t)(r + rbase) * DMODEL] = acc[t][r] + bv;
    }
  }
}

// ---------------------------------------------------------------------------
// Flash attention: one wave per (b, h, 16-query tile).
// Q,K bf16 [B,H,S,64]; Vt bf16 [B,H,64,S]; ctx bf16 [B,S,D].
// Scores arrive pre-scaled by log2e -> softmax runs on native v_exp (exp2).
__global__ __launch_bounds__(256) void flash_attn(
    const u16* __restrict__ Q, const u16* __restrict__ Kb, const u16* __restrict__ Vt,
    const unsigned char* __restrict__ amask, const int* __restrict__ is_causal_p,
    u16* __restrict__ ctx) {
  __shared__ u16 lds[8][16 * 32];                 // wave-private 16x32 bf16 P tile
  const int lane = threadIdx.x & 31;
  const int wslot = threadIdx.x >> 5;
  const int wid = blockIdx.x * 8 + wslot;
  const int qt = wid & 127;                       // S/16 tiles
  const int bh = wid >> 7;                        // b*H + h
  const int b = bh >> 4;
  const int h = bh & 15;
  const int q0 = qt * 16;
  const bool hi = lane >= 16;
  const int lm = lane & 15;
  const int rbase = hi ? 8 : 0;
  const int causal = (*is_causal_p) ? 1 : 0;

  const u16* Qh = Q + (size_t)bh * S_LEN * DK;
  const u16* Kh = Kb + (size_t)bh * S_LEN * DK;
  const u16* Vh = Vt + (size_t)bh * DK * S_LEN;

  // per-row mask row pointers (row fixed per r; column index varies per block)
  const unsigned char* mrowq[8];
#pragma unroll
  for (int r = 0; r < 8; ++r)
    mrowq[r] = amask + ((size_t)b * S_LEN + (q0 + r + rbase)) * S_LEN;

  // Q fragments for dk chunks [0,32) and [32,64)
  v16bf qf[2];
  {
    const u16* qrow = Qh + (size_t)(q0 + lm) * DK;
#pragma unroll
    for (int f = 0; f < 2; ++f) {
      BF16Frag fb;
      fb.q[0] = *(const v4u*)(qrow + f * 32 + (hi ? 16 : 0));
      fb.q[1] = *(const v4u*)(qrow + f * 32 + (hi ? 24 : 8));
      qf[f] = fb.v;
    }
  }
  const v16bf onesf = ones_bf16();

  float mrun[8], lrun[8];
  v8f o[4] = {};
#pragma unroll
  for (int r = 0; r < 8; ++r) { mrun[r] = NEG_BIG; lrun[r] = 0.f; }

  const int kend = causal ? (q0 + 16) : S_LEN;
  for (int kstart = 0; kstart < kend; kstart += 32) {
    // ---- prefetch next key block's K rows / V chunk (global_prefetch_b8)
    if (kstart + 32 < kend) {
      const int kn = kstart + 32;
#pragma unroll
      for (int t = 0; t < 2; ++t)
        __builtin_prefetch(Kh + (size_t)(kn + t * 16 + lm) * DK, 0, 3);
#pragma unroll
      for (int g = 0; g < 4; ++g)
        __builtin_prefetch(Vh + (size_t)(g * 16 + lm) * S_LEN + kn, 0, 3);
    }
    // ---- S = Q * K^T (two 16-key subtiles, dk reduced by 2 chained WMMAs)
    v8f sc[2];
#pragma unroll
    for (int t = 0; t < 2; ++t) {
      v8f a = {};
#pragma unroll
      for (int f = 0; f < 2; ++f) {
        const u16* krow = Kh + (size_t)(kstart + t * 16 + lm) * DK + f * 32 + (hi ? 16 : 0);
        BF16Frag kb_;
        kb_.q[0] = *(const v4u*)(krow);
        kb_.q[1] = *(const v4u*)(krow + 8);
        a = __builtin_amdgcn_wmma_f32_16x16x32_bf16(
            false, qf[f], false, kb_.v, (short)0, a, false, false);
      }
      sc[t] = a;
    }
    // ---- attention-mask (branchless byte loads, every block)
#pragma unroll
    for (int t = 0; t < 2; ++t) {
      const int kg = kstart + t * 16 + lm;
#pragma unroll
      for (int r = 0; r < 8; ++r) {
        const int mval = (int)mrowq[r][kg];
        sc[t][r] = mval ? NEG_BIG : sc[t][r];
      }
    }
    // ---- causal mask: only the diagonal block can be affected
    if (causal && (kstart + 32 > q0)) {
#pragma unroll
      for (int t = 0; t < 2; ++t) {
        const int kg = kstart + t * 16 + lm;
#pragma unroll
        for (int r = 0; r < 8; ++r) {
          const int qg = q0 + r + rbase;
          sc[t][r] = (kg > qg) ? NEG_BIG : sc[t][r];
        }
      }
    }
    // ---- row max (butterfly within 16-lane halves == along row)
    float mx[8];
#pragma unroll
    for (int r = 0; r < 8; ++r) mx[r] = fmaxf(sc[0][r], sc[1][r]);
#pragma unroll
    for (int r = 0; r < 8; ++r)
#pragma unroll
      for (int d = 1; d < 16; d <<= 1) mx[r] = fmaxf(mx[r], __shfl_xor(mx[r], d, 32));
    float alpha[8];
#pragma unroll
    for (int r = 0; r < 8; ++r) {
      const float mn = fmaxf(mrun[r], mx[r]);
      alpha[r] = fast_exp2(mrun[r] - mn);   // raw v_exp_f32
      mrun[r] = mn;
    }
    // ---- P = exp2(S - m)  (scores already in log2 domain)
#pragma unroll
    for (int r = 0; r < 8; ++r) {
      sc[0][r] = fast_exp2(sc[0][r] - mrun[r]);
      sc[1][r] = fast_exp2(sc[1][r] - mrun[r]);
    }
    // ---- rescale O
#pragma unroll
    for (int g = 0; g < 4; ++g)
#pragma unroll
      for (int r = 0; r < 8; ++r) o[g][r] *= alpha[r];

    // ---- stage P (16x32) to LDS as bf16, reload in A-fragment layout
    u16* pl = &lds[wslot][0];
#pragma unroll
    for (int t = 0; t < 2; ++t)
#pragma unroll
      for (int r = 0; r < 8; ++r)
        pl[(r + rbase) * 32 + t * 16 + lm] = f2bf(sc[t][r]);

    v16bf pf;
    {
      const u16* prow = pl + lm * 32;
      const int c0 = hi ? 16 : 0;
      BF16Frag fb;
      fb.q[0] = *(const v4u*)(prow + c0);
      fb.q[1] = *(const v4u*)(prow + c0 + 8);
      pf = fb.v;
    }
    // ---- row sums via WMMA: P(16x32) * ones(32x16) -> element r = rowsum(r)
    {
      v8f rsum = {};
      rsum = __builtin_amdgcn_wmma_f32_16x16x32_bf16(
          false, pf, false, onesf, (short)0, rsum, false, false);
#pragma unroll
      for (int r = 0; r < 8; ++r) lrun[r] = lrun[r] * alpha[r] + rsum[r];
    }
    // ---- O += P * V  (Vt rows are contiguous along keys)
#pragma unroll
    for (int g = 0; g < 4; ++g) {
      const u16* vrow = Vh + (size_t)(g * 16 + lm) * S_LEN + kstart + (hi ? 16 : 0);
      BF16Frag vb;
      vb.q[0] = *(const v4u*)(vrow);
      vb.q[1] = *(const v4u*)(vrow + 8);
      o[g] = __builtin_amdgcn_wmma_f32_16x16x32_bf16(
          false, pf, false, vb.v, (short)0, o[g], false, false);
    }
  }

  // ---- normalize and store context [B,S,D] bf16
  float inv[8];
#pragma unroll
  for (int r = 0; r < 8; ++r) inv[r] = (lrun[r] > 0.f) ? fast_rcp(lrun[r]) : 0.f;
#pragma unroll
  for (int g = 0; g < 4; ++g) {
    const int d = h * DK + g * 16 + lm;
    u16* base = ctx + ((size_t)b * S_LEN + q0) * DMODEL + d;
#pragma unroll
    for (int r = 0; r < 8; ++r)
      base[(size_t)(r + rbase) * DMODEL] = f2bf(o[g][r] * inv[r]);
  }
}

// ---------------------------------------------------------------------------
extern "C" void kernel_launch(void* const* d_in, const int* in_sizes, int n_in,
                              void* d_out, int out_size, void* d_ws, size_t ws_size,
                              hipStream_t stream) {
  const float* query = (const float*)d_in[0];
  const float* key   = (const float*)d_in[1];
  const float* value = (const float*)d_in[2];
  const unsigned char* amask = (const unsigned char*)d_in[3];
  const float* Wq   = (const float*)d_in[4];
  const float* Wk   = (const float*)d_in[5];
  const float* Wv   = (const float*)d_in[6];
  const float* Wout = (const float*)d_in[7];
  const float* bout = (const float*)d_in[8];
  const int* isc    = (const int*)d_in[9];
  float* out = (float*)d_out;

  const int WN = DMODEL * DMODEL;                 // 1048576
  const int QN = BATCH * S_LEN * DMODEL;          // 4194304

  char* ws = (char*)d_ws;
  size_t off = 0;
  auto alloc = [&](size_t bytes) -> void* {
    void* p = ws + off;
    off += (bytes + 255) & ~(size_t)255;
    return p;
  };
  u16* wq_bf = (u16*)alloc((size_t)WN * 2);
  u16* wk_bf = (u16*)alloc((size_t)WN * 2);
  u16* wv_bf = (u16*)alloc((size_t)WN * 2);
  u16* wo_bf = (u16*)alloc((size_t)WN * 2);
  u16* qx_bf = (u16*)alloc((size_t)QN * 2);       // query bf16
  u16* kx_bf = (u16*)alloc((size_t)QN * 2);       // key bf16
  u16* vx_bf = (u16*)alloc((size_t)QN * 2);       // value bf16
  u16* qbuf  = (u16*)alloc((size_t)QN * 2);       // Q [B,H,S,dk]
  u16* kbuf  = (u16*)alloc((size_t)QN * 2);       // K [B,H,S,dk]
  u16* vtbuf = (u16*)alloc((size_t)QN * 2);       // V^T [B,H,dk,S]
  u16* ctx   = (u16*)alloc((size_t)QN * 2);       // context bf16  (~67 MB total)

  cvt_f32_bf16<<<256, 256, 0, stream>>>(Wq, wq_bf, WN / 8);
  cvt_f32_bf16<<<256, 256, 0, stream>>>(Wk, wk_bf, WN / 8);
  cvt_f32_bf16<<<256, 256, 0, stream>>>(Wv, wv_bf, WN / 8);
  cvt_f32_bf16<<<256, 256, 0, stream>>>(Wout, wo_bf, WN / 8);
  cvt_f32_bf16<<<1024, 256, 0, stream>>>(query, qx_bf, QN / 8);
  cvt_f32_bf16<<<1024, 256, 0, stream>>>(key,   kx_bf, QN / 8);
  cvt_f32_bf16<<<1024, 256, 0, stream>>>(value, vx_bf, QN / 8);

  // 4096 waves each (M/16 * N/64 = 256*16), 8 waves/block
  gemm_xwt<0><<<512, 256, 0, stream>>>(qx_bf, wq_bf, qbuf, nullptr, nullptr);
  gemm_xwt<1><<<512, 256, 0, stream>>>(kx_bf, wk_bf, kbuf, nullptr, nullptr);
  gemm_xwt<2><<<512, 256, 0, stream>>>(vx_bf, wv_bf, vtbuf, nullptr, nullptr);

  // B*H*(S/16) = 4096 waves
  flash_attn<<<512, 256, 0, stream>>>(qbuf, kbuf, vtbuf, amask, isc, ctx);

  gemm_xwt<3><<<512, 256, 0, stream>>>(ctx, wo_bf, nullptr, out, bout);
}
